// MultiHeadAttention_446676599424
// MI455X (gfx1250) — matrix operand order repaired and checked by reading, not compiled, not run on hardware
//
#include <hip/hip_runtime.h>
#include <hip/hip_bf16.h>

typedef unsigned short u16;
typedef __attribute__((ext_vector_type(16))) __bf16   v16bf;
typedef __attribute__((ext_vector_type(8)))  float    v8f;
typedef __attribute__((ext_vector_type(4)))  unsigned v4u;

#define D_MODEL 1024
#define N_HEAD  16
#define D_K     64
#define SEQ     2048
#define BATCH   2

union Frag { v4u q[2]; v16bf v; };

__device__ __forceinline__ unsigned bf16r(float f) {
    unsigned u = __builtin_bit_cast(unsigned, f);
    return (u + 0x7FFFu + ((u >> 16) & 1u)) >> 16;   // round-to-nearest-even
}
__device__ __forceinline__ unsigned pkbf(float lo, float hi) {
    return bf16r(lo) | (bf16r(hi) << 16);
}

// f32 -> bf16 conversion (one-shot, memory bound)
__global__ void cvt_f32_bf16(const float* __restrict__ in, u16* __restrict__ out, int n)
{
    int i = blockIdx.x * blockDim.x + threadIdx.x;
    if (i < n) out[i] = (u16)bf16r(in[i]);
}

// Batched NT GEMM: C[M,N] = A[M,K] * B[N,K]^T, bf16 WMMA, f32 accumulate.
// B is always bf16. A is bf16 or f32 (AF32). C is stored f32 (CF32) or bf16.
// Each wave computes a 32x32 C tile (2x2 WMMA sub-tiles -> 4 wmma per K-step).
// grid.y = batch (b*H+h) with independent strides; grid.x covers (M/32)*(N/32) wave tiles.
template<bool AF32, bool CF32>
__global__ void gemm_nt_wmma(const void* __restrict__ A_, const u16* __restrict__ B,
                             void* __restrict__ C_,
                             int lda, int ldb, int ldc, int K, int tilesN, int H,
                             long long aOffB, long long aOffH,
                             long long bOffB, long long bOffH,
                             long long cOffB, long long cOffH)
{
    const int batch = blockIdx.y;
    const int b = batch / H;
    const int h = batch % H;

    const int wave = blockIdx.x * (blockDim.x >> 5) + (threadIdx.x >> 5);
    const int lane = threadIdx.x & 31;
    const int half = lane >> 4;      // 0: lanes 0-15, 1: lanes 16-31
    const int l    = lane & 15;

    const int tm = wave / tilesN;
    const int tn = wave % tilesN;
    const int m0 = tm << 5;
    const int n0 = tn << 5;

    const u16* __restrict__ Bb = B + (size_t)b * bOffB + (size_t)h * bOffH;
    const u16* __restrict__ brow0 = Bb + (size_t)(n0      + l) * ldb + half * 16;
    const u16* __restrict__ brow1 = Bb + (size_t)(n0 + 16 + l) * ldb + half * 16;

    v8f acc00 = {0.f,0.f,0.f,0.f,0.f,0.f,0.f,0.f};
    v8f acc01 = acc00, acc10 = acc00, acc11 = acc00;

    if constexpr (AF32) {
        const float* Ab = (const float*)A_ + (size_t)b * aOffB + (size_t)h * aOffH;
        const float* arow0 = Ab + (size_t)(m0      + l) * lda + half * 16;
        const float* arow1 = Ab + (size_t)(m0 + 16 + l) * lda + half * 16;
        for (int k0 = 0; k0 < K; k0 += 32) {
            __builtin_prefetch(arow0 + k0 + 128, 0, 0);
            __builtin_prefetch(arow1 + k0 + 128, 0, 0);
            Frag fa0, fa1;
            #pragma unroll
            for (int i = 0; i < 4; ++i) {
                fa0.q[0][i] = pkbf(arow0[k0 + 2*i],      arow0[k0 + 2*i + 1]);
                fa0.q[1][i] = pkbf(arow0[k0 + 8 + 2*i], arow0[k0 + 8 + 2*i + 1]);
                fa1.q[0][i] = pkbf(arow1[k0 + 2*i],      arow1[k0 + 2*i + 1]);
                fa1.q[1][i] = pkbf(arow1[k0 + 8 + 2*i], arow1[k0 + 8 + 2*i + 1]);
            }
            v16bf bv0 = *(const v16bf*)(brow0 + k0);
            v16bf bv1 = *(const v16bf*)(brow1 + k0);
            acc00 = __builtin_amdgcn_wmma_f32_16x16x32_bf16(false, fa0.v, false, bv0, (short)0, acc00, false, false);
            acc01 = __builtin_amdgcn_wmma_f32_16x16x32_bf16(false, fa0.v, false, bv1, (short)0, acc01, false, false);
            acc10 = __builtin_amdgcn_wmma_f32_16x16x32_bf16(false, fa1.v, false, bv0, (short)0, acc10, false, false);
            acc11 = __builtin_amdgcn_wmma_f32_16x16x32_bf16(false, fa1.v, false, bv1, (short)0, acc11, false, false);
        }
    } else {
        const u16* Ab = (const u16*)A_ + (size_t)b * aOffB + (size_t)h * aOffH;
        const u16* arow0 = Ab + (size_t)(m0      + l) * lda + half * 16;
        const u16* arow1 = Ab + (size_t)(m0 + 16 + l) * lda + half * 16;
        for (int k0 = 0; k0 < K; k0 += 32) {
            __builtin_prefetch(arow0 + k0 + 256, 0, 0);
            __builtin_prefetch(brow0 + k0 + 256, 0, 0);
            Frag fa0, fa1;
            fa0.q[0] = *(const v4u*)(arow0 + k0);
            fa0.q[1] = *(const v4u*)(arow0 + k0 + 8);
            fa1.q[0] = *(const v4u*)(arow1 + k0);
            fa1.q[1] = *(const v4u*)(arow1 + k0 + 8);
            v16bf bv0 = *(const v16bf*)(brow0 + k0);
            v16bf bv1 = *(const v16bf*)(brow1 + k0);
            acc00 = __builtin_amdgcn_wmma_f32_16x16x32_bf16(false, fa0.v, false, bv0, (short)0, acc00, false, false);
            acc01 = __builtin_amdgcn_wmma_f32_16x16x32_bf16(false, fa0.v, false, bv1, (short)0, acc01, false, false);
            acc10 = __builtin_amdgcn_wmma_f32_16x16x32_bf16(false, fa1.v, false, bv0, (short)0, acc10, false, false);
            acc11 = __builtin_amdgcn_wmma_f32_16x16x32_bf16(false, fa1.v, false, bv1, (short)0, acc11, false, false);
        }
    }

    // Epilogue: C layout VGPR r -> row m0 + mi*16 + half*8 + r, col n0 + ni*16 + l
    if constexpr (CF32) {
        float* Cb = (float*)C_ + (size_t)b * cOffB + (size_t)h * cOffH;
        #pragma unroll
        for (int r = 0; r < 8; ++r) {
            size_t r0 = (size_t)(m0 + half * 8 + r) * ldc + n0 + l;
            size_t r1 = (size_t)(m0 + 16 + half * 8 + r) * ldc + n0 + l;
            Cb[r0]      = acc00[r];
            Cb[r0 + 16] = acc01[r];
            Cb[r1]      = acc10[r];
            Cb[r1 + 16] = acc11[r];
        }
    } else {
        u16* Cb = (u16*)C_ + (size_t)b * cOffB + (size_t)h * cOffH;
        #pragma unroll
        for (int r = 0; r < 8; ++r) {
            size_t r0 = (size_t)(m0 + half * 8 + r) * ldc + n0 + l;
            size_t r1 = (size_t)(m0 + 16 + half * 8 + r) * ldc + n0 + l;
            Cb[r0]      = (u16)bf16r(acc00[r]);
            Cb[r0 + 16] = (u16)bf16r(acc01[r]);
            Cb[r1]      = (u16)bf16r(acc10[r]);
            Cb[r1 + 16] = (u16)bf16r(acc11[r]);
        }
    }
}

// vT[b, h, d, s] = v[b, s, h*D_K + d]   (bf16)
__global__ void transpose_v(const u16* __restrict__ v, u16* __restrict__ vT)
{
    size_t idx = (size_t)blockIdx.x * blockDim.x + threadIdx.x; // over B*H*D_K*SEQ
    int s = (int)(idx % SEQ);  size_t t = idx / SEQ;
    int d = (int)(t % D_K);    t /= D_K;
    int h = (int)(t % N_HEAD);
    int b = (int)(t / N_HEAD);
    vT[idx] = v[((size_t)b * SEQ + s) * D_MODEL + h * D_K + d];
}

// Row softmax (2048 cols) with 1/sqrt(D_K) scale folded in; in place on f32.
__global__ void softmax_rows(float* __restrict__ att)
{
    __shared__ float sm[256];
    const int tid = threadIdx.x;
    float* __restrict__ row = att + (size_t)blockIdx.x * SEQ;

    float vals[8];
    float vmax = -3.4e38f;
    #pragma unroll
    for (int i = 0; i < 8; ++i) {
        vals[i] = row[tid + i * 256] * 0.125f;
        vmax = fmaxf(vmax, vals[i]);
    }
    sm[tid] = vmax; __syncthreads();
    for (int s = 128; s > 0; s >>= 1) { if (tid < s) sm[tid] = fmaxf(sm[tid], sm[tid + s]); __syncthreads(); }
    const float m = sm[0]; __syncthreads();

    float lsum = 0.f;
    #pragma unroll
    for (int i = 0; i < 8; ++i) { vals[i] = expf(vals[i] - m); lsum += vals[i]; }
    sm[tid] = lsum; __syncthreads();
    for (int s = 128; s > 0; s >>= 1) { if (tid < s) sm[tid] += sm[tid + s]; __syncthreads(); }
    const float inv = 1.f / sm[0];

    #pragma unroll
    for (int i = 0; i < 8; ++i) row[tid + i * 256] = vals[i] * inv;
}

// out = LayerNorm(attn_out + residual), last dim 1024. One block per row.
__global__ void add_layernorm(const float* __restrict__ attn_out,
                              const float* __restrict__ resid,
                              const float* __restrict__ gamma,
                              const float* __restrict__ beta,
                              float* __restrict__ out)
{
    __shared__ float sm[256];
    const int tid = threadIdx.x;
    const size_t base = (size_t)blockIdx.x * D_MODEL;

    float x[4];
    float lsum = 0.f;
    #pragma unroll
    for (int i = 0; i < 4; ++i) {
        int c = tid + i * 256;
        x[i] = attn_out[base + c] + resid[base + c];
        lsum += x[i];
    }
    sm[tid] = lsum; __syncthreads();
    for (int s = 128; s > 0; s >>= 1) { if (tid < s) sm[tid] += sm[tid + s]; __syncthreads(); }
    const float mu = sm[0] * (1.f / D_MODEL); __syncthreads();

    float lsq = 0.f;
    #pragma unroll
    for (int i = 0; i < 4; ++i) { float d = x[i] - mu; lsq += d * d; }
    sm[tid] = lsq; __syncthreads();
    for (int s = 128; s > 0; s >>= 1) { if (tid < s) sm[tid] += sm[tid + s]; __syncthreads(); }
    const float rstd = rsqrtf(sm[0] * (1.f / D_MODEL) + 1e-5f);

    #pragma unroll
    for (int i = 0; i < 4; ++i) {
        int c = tid + i * 256;
        out[base + c] = (x[i] - mu) * rstd * gamma[c] + beta[c];
    }
}

extern "C" void kernel_launch(void* const* d_in, const int* in_sizes, int n_in,
                              void* d_out, int out_size, void* d_ws, size_t ws_size,
                              hipStream_t stream)
{
    (void)in_sizes; (void)n_in; (void)out_size; (void)ws_size;
    const float* Qin   = (const float*)d_in[0];
    const float* Kin   = (const float*)d_in[1];
    const float* Vin   = (const float*)d_in[2];
    const float* W_Q   = (const float*)d_in[3];
    const float* W_K   = (const float*)d_in[4];
    const float* W_V   = (const float*)d_in[5];
    const float* W_O   = (const float*)d_in[6];
    const float* gamma = (const float*)d_in[7];
    const float* beta  = (const float*)d_in[8];

    float* out  = (float*)d_out;                           // [B,S,D]
    float* attn = out + (size_t)BATCH * SEQ * D_MODEL;     // [B,H,S,S] f32 (returned)

    const size_t F = (size_t)BATCH * SEQ * D_MODEL;        // 4,194,304 elems
    const size_t W = (size_t)D_MODEL * D_MODEL;            // 1,048,576 elems

    u16* U    = (u16*)d_ws;
    u16* qb   = U;                 // bf16 q proj           [0,   F)
    u16* kb   = U + F;             // bf16 k proj           [F,  2F)
    u16* vb   = U + 2 * F;         // bf16 v proj  -> ctx reuse
    u16* vT   = U + 3 * F;         // bf16 v transposed [B,H,D_K,S]
    u16* Qbf  = U + 4 * F;         // bf16 copies of inputs
    u16* Kbf  = U + 5 * F;
    u16* Vbf  = U + 6 * F;
    u16* Wqb  = U + 7 * F;
    u16* Wkb  = Wqb + W;
    u16* Wvb  = Wqb + 2 * W;
    u16* Wob  = Wqb + 3 * W;
    u16* ctx  = vb;                       // v dead after transpose
    float* aout = (float*)U;              // f32, reuses qb+kb (dead after scores)

    const int M = BATCH * SEQ;            // 4096

    // 0: one-shot bf16 conversions
    cvt_f32_bf16<<<(int)(F / 256), 256, 0, stream>>>(Qin, Qbf, (int)F);
    cvt_f32_bf16<<<(int)(F / 256), 256, 0, stream>>>(Kin, Kbf, (int)F);
    cvt_f32_bf16<<<(int)(F / 256), 256, 0, stream>>>(Vin, Vbf, (int)F);
    cvt_f32_bf16<<<(int)(W / 256), 256, 0, stream>>>(W_Q, Wqb, (int)W);
    cvt_f32_bf16<<<(int)(W / 256), 256, 0, stream>>>(W_K, Wkb, (int)W);
    cvt_f32_bf16<<<(int)(W / 256), 256, 0, stream>>>(W_V, Wvb, (int)W);
    cvt_f32_bf16<<<(int)(W / 256), 256, 0, stream>>>(W_O, Wob, (int)W);

    // 1-3: projections [4096,1024] x [1024,1024]^T -> bf16
    {
        dim3 grid((M / 32) * (D_MODEL / 32) / 4, 1);
        gemm_nt_wmma<false, false><<<grid, 128, 0, stream>>>(Qbf, Wqb, qb,
            D_MODEL, D_MODEL, D_MODEL, D_MODEL, D_MODEL / 32, 1, 0, 0, 0, 0, 0, 0);
        gemm_nt_wmma<false, false><<<grid, 128, 0, stream>>>(Kbf, Wkb, kb,
            D_MODEL, D_MODEL, D_MODEL, D_MODEL, D_MODEL / 32, 1, 0, 0, 0, 0, 0, 0);
        gemm_nt_wmma<false, false><<<grid, 128, 0, stream>>>(Vbf, Wvb, vb,
            D_MODEL, D_MODEL, D_MODEL, D_MODEL, D_MODEL / 32, 1, 0, 0, 0, 0, 0, 0);
    }

    // 4: transpose V -> [B,H,D_K,S]
    transpose_v<<<(int)(F / 256), 256, 0, stream>>>(vb, vT);

    // 5: scores[b,h] = q_h [S,64] x k_h [S,64]^T -> f32 in d_out
    {
        dim3 grid((SEQ / 32) * (SEQ / 32) / 4, BATCH * N_HEAD);
        gemm_nt_wmma<false, true><<<grid, 128, 0, stream>>>(
            qb, kb, attn, D_MODEL, D_MODEL, SEQ, D_K, SEQ / 32, N_HEAD,
            (long long)SEQ * D_MODEL, D_K,
            (long long)SEQ * D_MODEL, D_K,
            (long long)N_HEAD * SEQ * SEQ, (long long)SEQ * SEQ);
    }

    // 6: softmax (applies 1/8 scale), in place on f32 attention weights
    softmax_rows<<<BATCH * N_HEAD * SEQ, 256, 0, stream>>>(attn);

    // 7: context[b,h] = P [S,S](f32) x vT_h [64,S]^T(bf16) -> bf16 [B,S,D] slices
    {
        dim3 grid((SEQ / 32) * (D_K / 32) / 4, BATCH * N_HEAD);
        gemm_nt_wmma<true, false><<<grid, 128, 0, stream>>>(
            attn, vT, ctx, SEQ, SEQ, D_MODEL, SEQ, D_K / 32, N_HEAD,
            (long long)N_HEAD * SEQ * SEQ, (long long)SEQ * SEQ,
            (long long)N_HEAD * D_K * SEQ, (long long)D_K * SEQ,
            (long long)SEQ * D_MODEL, D_K);
    }

    // 8: output projection: ctx [4096,1024] x W_O^T -> f32
    {
        dim3 grid((M / 32) * (D_MODEL / 32) / 4, 1);
        gemm_nt_wmma<false, true><<<grid, 128, 0, stream>>>(ctx, Wob, aout,
            D_MODEL, D_MODEL, D_MODEL, D_MODEL, D_MODEL / 32, 1, 0, 0, 0, 0, 0, 0);
    }

    // 9: residual + LayerNorm
    add_layernorm<<<M, 256, 0, stream>>>(aout, Qin, gamma, beta, out);
}
